// Llama4TextMoeEP_1460288880660
// MI455X (gfx1250) — matrix operand order repaired and checked by reading, not compiled, not run on hardware
//
#include <hip/hip_runtime.h>
#include <hip/hip_bf16.h>
#include <math.h>

#define TOK 4096
#define HID 2048
#define FDIM 4096
#define NE 8

typedef __attribute__((ext_vector_type(8)))  float  v8f;
typedef __attribute__((ext_vector_type(16))) __bf16 v16bf;

#if defined(__gfx1250__) && __has_builtin(__builtin_amdgcn_global_load_async_to_lds_b128)
#define HAVE_ASYNC_LDS 1
#else
#define HAVE_ASYNC_LDS 0
#endif

// RNE f32->bf16 (epilogue only).
__device__ __forceinline__ unsigned short f2bf(float f) {
  unsigned u = __builtin_bit_cast(unsigned, f);
  unsigned r = u + 0x7FFFu + ((u >> 16) & 1u);
  return (unsigned short)(r >> 16);
}

// 1-op truncating f32->bf16 for scattered stores.
__device__ __forceinline__ unsigned short f2bf_fast(float f) {
  return (unsigned short)(__builtin_bit_cast(unsigned, f) >> 16);
}

// Pack two f32 -> bf16x2 in one op. Prefer v_cvt_pk_bf16_f32 (RNE) when the
// target has it; else a single v_perm_b32 picking the high halves:
// result bytes [a.b2, a.b3, b.b2, b.b3]  => selector 0x03020706 (S0=a hi region).
__device__ __forceinline__ unsigned pack2bf(float a, float b) {
#if defined(__gfx1250__) && __has_builtin(__builtin_amdgcn_cvt_pk_bf16_f32)
  typedef __attribute__((ext_vector_type(2))) __bf16 v2bf;
  v2bf r = __builtin_amdgcn_cvt_pk_bf16_f32(a, b);
  return __builtin_bit_cast(unsigned, r);
#else
  return __builtin_amdgcn_perm(__builtin_bit_cast(unsigned, a),
                               __builtin_bit_cast(unsigned, b), 0x03020706u);
#endif
}

#if HAVE_ASYNC_LDS
// Builtin signature (probed): (v4i __device__* gsrc, v4i __shared__* ldst, Ii, Ii).
typedef int v4i_vs __attribute__((vector_size(16)));
typedef __attribute__((address_space(1))) v4i_vs gv4i;
typedef __attribute__((address_space(3))) v4i_vs lv4i;
__device__ __forceinline__ void async_b128(const void* g, const unsigned short* l) {
  __builtin_amdgcn_global_load_async_to_lds_b128(
      (gv4i*)(unsigned long long)(size_t)g,
      (lv4i*)(unsigned)(size_t)l, 0, 0);
}
__device__ __forceinline__ void wait_async0() {
#if __has_builtin(__builtin_amdgcn_s_wait_asynccnt)
  __builtin_amdgcn_s_wait_asynccnt(0);
#else
  asm volatile("s_wait_asynccnt 0" ::: "memory");
#endif
}
#endif

__device__ __forceinline__ v8f vzero() {
  v8f v;
#pragma unroll
  for (int i = 0; i < 8; ++i) v[i] = 0.0f;
  return v;
}

// A fragment: 16x32 bf16, LDS [m][k], k contiguous (stride 32 elems).
__device__ __forceinline__ v16bf frag_a(const unsigned short* lds, int row_base, int lane) {
  union { v16bf v; unsigned u[8]; } f;
  const unsigned* p = (const unsigned*)(lds + (row_base + (lane & 15)) * 32);
  const int kh2 = (lane < 16) ? 0 : 4;
#pragma unroll
  for (int i = 0; i < 4; ++i) f.u[i] = p[kh2 + i];
#pragma unroll
  for (int i = 4; i < 8; ++i) f.u[i] = p[kh2 + i + 4];
  return f.v;
}

// B fragment: 32x16 bf16, LDS [n][k], k contiguous (stride 32 elems).
__device__ __forceinline__ v16bf frag_b(const unsigned short* lds, int col_base, int lane) {
  union { v16bf v; unsigned u[8]; } f;
  const unsigned* p =
      (const unsigned*)(lds + (col_base + (lane & 15)) * 32 + ((lane < 16) ? 0 : 8));
#pragma unroll
  for (int i = 0; i < 8; ++i) f.u[i] = p[i < 4 ? i : i + 4];
  return f.v;
}

#define WMMA_BF16(a, b, c) \
  __builtin_amdgcn_wmma_f32_16x16x32_bf16(false, (a), false, (b), (short)0, (c), false, false)

// ---------------------------------------------------------------------------
// 1) Router: one wave per token.
// ---------------------------------------------------------------------------
__global__ __launch_bounds__(32)
void router_kernel(const float* __restrict__ X, const float* __restrict__ Wr,
                   float* __restrict__ logits_out, int* __restrict__ expert_id,
                   float* __restrict__ score) {
  const int t = blockIdx.x;
  const int lane = threadIdx.x;
  float acc[NE];
#pragma unroll
  for (int e = 0; e < NE; ++e) acc[e] = 0.0f;
  const float* x = X + (size_t)t * HID;
  for (int h = lane; h < HID; h += 32) {
    const float xv = x[h];
#pragma unroll
    for (int e = 0; e < NE; ++e) acc[e] += xv * Wr[e * HID + h];
  }
#pragma unroll
  for (int e = 0; e < NE; ++e) {
#pragma unroll
    for (int off = 16; off > 0; off >>= 1) acc[e] += __shfl_xor(acc[e], off, 32);
  }
  if (lane == 0) {
    int best = 0;
    float bv = acc[0];
#pragma unroll
    for (int e = 1; e < NE; ++e) {
      if (acc[e] > bv) { bv = acc[e]; best = e; }  // strict >: first-max tie-break
    }
    expert_id[t] = best;
    score[t] = 1.0f / (1.0f + __expf(-bv));
#pragma unroll
    for (int e = 0; e < NE; ++e) logits_out[(size_t)t * NE + e] = acc[e];
  }
}

// ---------------------------------------------------------------------------
// 2) Deterministic counts/offsets + stable token ordering.
// ---------------------------------------------------------------------------
__global__ __launch_bounds__(256)
void count_kernel(const int* __restrict__ expert_id, int* __restrict__ counts,
                  int* __restrict__ offsets) {
  __shared__ int c[NE];
  if (threadIdx.x < NE) c[threadIdx.x] = 0;
  __syncthreads();
  for (int t = threadIdx.x; t < TOK; t += blockDim.x) atomicAdd(&c[expert_id[t]], 1);
  __syncthreads();
  if (threadIdx.x == 0) {
    int run = 0;
#pragma unroll
    for (int e = 0; e < NE; ++e) {
      offsets[e] = run;
      counts[e] = c[e];
      run += c[e];
    }
  }
}

// Stable rank (slot = offsets[e] + #earlier tokens with same expert), with
// expert ids staged through LDS so the O(T) scan never touches global memory.
__global__ __launch_bounds__(256)
void rank_kernel(const int* __restrict__ expert_id, const int* __restrict__ offsets,
                 int* __restrict__ tlist) {
  __shared__ int eid[TOK];
  for (int i = threadIdx.x; i < TOK; i += blockDim.x) eid[i] = expert_id[i];
  __syncthreads();
  const int t = blockIdx.x * blockDim.x + threadIdx.x;
  if (t >= TOK) return;
  const int e = eid[t];
  int r = 0;
  for (int i = 0; i < t; ++i) r += (eid[i] == e) ? 1 : 0;
  tlist[offsets[e] + r] = t;
}

// ---------------------------------------------------------------------------
// 3) Fused gate_up GEMM + SwiGLU. Tile 128m x 64f x K32, double-buffered LDS,
//    register prefetch of next tile so global loads overlap the WMMAs.
// ---------------------------------------------------------------------------
template <bool ROUTED>
__global__ __launch_bounds__(256)
void moe_gateup(const float* __restrict__ X, const float* __restrict__ Wg0,
                const float* __restrict__ Wu0, const int* __restrict__ counts,
                const int* __restrict__ offsets, const int* __restrict__ tlist,
                const float* __restrict__ score, unsigned short* __restrict__ inter) {
  __shared__ __align__(16) unsigned short As[2 * 128 * 32];
  __shared__ __align__(16) unsigned short Bg[2 * 64 * 32];
  __shared__ __align__(16) unsigned short Bu[2 * 64 * 32];

  const int tid = threadIdx.x;
  const int e = blockIdx.z;
  const int cnt = ROUTED ? counts[e] : TOK;
  const int base = ROUTED ? offsets[e] : 0;
  const int m0 = blockIdx.x * 128;   // m fastest: concurrent blocks share B tiles in L2
  if (m0 >= cnt) return;
  const int f0 = blockIdx.y * 64;

  const float* Wg;
  const float* Wu;
  long ldb;
  if (ROUTED) {
    const float* wbase = Wg0 + (size_t)e * HID * (2 * FDIM);
    Wg = wbase;
    Wu = wbase + FDIM;
    ldb = 2 * FDIM;
  } else {
    Wg = Wg0;
    Wu = Wu0;
    ldb = HID;
  }

  // A: thread -> (row, k-half)
  const int arow = tid >> 1;
  const int akh = (tid & 1) * 16;
  const int aslot = m0 + arow;
  const bool avalid = (aslot < cnt);
  int atok = 0;
  float ascale = 0.0f;
  if (avalid) {
    atok = ROUTED ? tlist[base + aslot] : aslot;
    ascale = ROUTED ? score[atok] : 1.0f;
  }
  const float* aSrc = avalid ? (X + (size_t)atok * HID + akh) : X;

  // B: thread mapping (tens: 0=gate, 1=up)
  const int tens = tid >> 7;
  const int br = tid & 127;
  const int bk = br >> 2;            // ROUTED: k row 0..31
  const int bn0 = (br & 3) * 16;     // ROUTED: n start
  const int bn = br >> 1;            // shared: n row 0..63
  const int bkh = (br & 1) * 16;     // shared: k half
  const float* Wsel = tens ? Wu : Wg;

  v8f accG[4], accU[4];
#pragma unroll
  for (int i = 0; i < 4; ++i) { accG[i] = vzero(); accU[i] = vzero(); }

  const int wave = tid >> 5, lane = tid & 31;
  const int wm = wave & 3, wn = wave >> 2;

  float aR[16], bR[16];
  auto loadA = [&](int k0) {
    if (avalid) {
      const float* s = aSrc + k0;
#pragma unroll
      for (int i = 0; i < 16; ++i) aR[i] = s[i];
    } else {
#pragma unroll
      for (int i = 0; i < 16; ++i) aR[i] = 0.0f;
    }
  };
  auto loadB = [&](int k0) {
    const float* s = ROUTED ? (Wsel + (size_t)(k0 + bk) * ldb + f0 + bn0)
                            : (Wsel + (size_t)(f0 + bn) * ldb + k0 + bkh);
#pragma unroll
    for (int i = 0; i < 16; ++i) bR[i] = s[i];
  };
  auto storeA = [&](int buf) {
    union { unsigned u[8]; uint4 q[2]; } t;
#pragma unroll
    for (int j = 0; j < 8; ++j) t.u[j] = pack2bf(aR[2 * j] * ascale, aR[2 * j + 1] * ascale);
    uint4* dst = (uint4*)(As + buf * (128 * 32) + arow * 32 + akh);
    dst[0] = t.q[0];
    dst[1] = t.q[1];
  };
  auto storeB = [&](int buf) {
    unsigned short* BL = (tens ? Bu : Bg) + buf * (64 * 32);
    if (ROUTED) {
#pragma unroll
      for (int i = 0; i < 16; ++i) BL[(bn0 + i) * 32 + bk] = f2bf_fast(bR[i]);  // [n][k] scatter
    } else {
      union { unsigned u[8]; uint4 q[2]; } t;
#pragma unroll
      for (int j = 0; j < 8; ++j) t.u[j] = pack2bf(bR[2 * j], bR[2 * j + 1]);
      uint4* dst = (uint4*)(BL + bn * 32 + bkh);
      dst[0] = t.q[0];
      dst[1] = t.q[1];
    }
  };

  const int NIT = HID / 32;
  loadA(0);
  loadB(0);
  for (int i = 0; i < NIT; ++i) {
    const int buf = i & 1;
    storeA(buf);
    storeB(buf);
    __syncthreads();  // publish tile i (single barrier per step: double-buffered)
    if (i + 1 < NIT) {  // prefetch tile i+1 while WMMAs run
      loadA((i + 1) * 32);
      loadB((i + 1) * 32);
    }
    const unsigned short* Ab = As + buf * (128 * 32);
    const unsigned short* Bgb = Bg + buf * (64 * 32);
    const unsigned short* Bub = Bu + buf * (64 * 32);
    v16bf a0 = frag_a(Ab, wm * 32, lane);
    v16bf a1 = frag_a(Ab, wm * 32 + 16, lane);
    v16bf bg0 = frag_b(Bgb, wn * 32, lane);
    v16bf bg1 = frag_b(Bgb, wn * 32 + 16, lane);
    v16bf bu0 = frag_b(Bub, wn * 32, lane);
    v16bf bu1 = frag_b(Bub, wn * 32 + 16, lane);
    accG[0] = WMMA_BF16(a0, bg0, accG[0]);
    accG[1] = WMMA_BF16(a0, bg1, accG[1]);
    accG[2] = WMMA_BF16(a1, bg0, accG[2]);
    accG[3] = WMMA_BF16(a1, bg1, accG[3]);
    accU[0] = WMMA_BF16(a0, bu0, accU[0]);
    accU[1] = WMMA_BF16(a0, bu1, accU[1]);
    accU[2] = WMMA_BF16(a1, bu0, accU[2]);
    accU[3] = WMMA_BF16(a1, bu1, accU[3]);
  }

  // Epilogue: act = up * silu(gate); C layout: n=lane&15, m=(lane<16?0:8)+reg.
  const int nloc = lane & 15;
  const int mb = (lane < 16) ? 0 : 8;
#pragma unroll
  for (int mi = 0; mi < 2; ++mi) {
#pragma unroll
    for (int ni = 0; ni < 2; ++ni) {
      v8f g = accG[mi * 2 + ni];
      v8f u = accU[mi * 2 + ni];
#pragma unroll
      for (int r = 0; r < 8; ++r) {
        const int m = wm * 32 + mi * 16 + mb + r;
        const int slot = m0 + m;
        if (slot < cnt) {
          const float gv = g[r];
          const float act = u[r] * gv / (1.0f + __expf(-gv));
          const int n = f0 + wn * 32 + ni * 16 + nloc;
          inter[(size_t)(base + slot) * FDIM + n] = f2bf(act);
        }
      }
    }
  }
}

// ---------------------------------------------------------------------------
// 4) Down GEMM. Tile 128m x 128h x K32 over F, double-buffered; A tile is raw
//    bf16 so it rides the async global->LDS DMA path when available.
// ---------------------------------------------------------------------------
template <bool ROUTED>
__global__ __launch_bounds__(256)
void moe_down(const unsigned short* __restrict__ inter, const float* __restrict__ Wd0,
              const int* __restrict__ counts, const int* __restrict__ offsets,
              const int* __restrict__ tlist, float* __restrict__ out) {
  __shared__ __align__(16) unsigned short As[2 * 128 * 32];
  __shared__ __align__(16) unsigned short Bs[2 * 128 * 32];

  const int tid = threadIdx.x;
  const int e = blockIdx.z;
  const int cnt = ROUTED ? counts[e] : TOK;
  const int base = ROUTED ? offsets[e] : 0;
  const int m0 = blockIdx.x * 128;
  if (m0 >= cnt) return;
  const int h0 = blockIdx.y * 128;
  const float* Wd = ROUTED ? (Wd0 + (size_t)e * FDIM * HID) : Wd0;

  const int arow = tid >> 1, akh = (tid & 1) * 16;
  const int aslot = m0 + arow;
  const bool avalid = (aslot < cnt);
  const unsigned short* aSrc = inter + (size_t)(base + (avalid ? aslot : 0)) * FDIM + akh;

  const int bk = tid >> 3;          // ROUTED: k row
  const int bn0 = (tid & 7) * 16;   // ROUTED: n start
  const int bn = tid >> 1;          // shared: n row
  const int bkh = (tid & 1) * 16;   // shared: k half

  v8f acc[8];
#pragma unroll
  for (int i = 0; i < 8; ++i) acc[i] = vzero();

  const int wave = tid >> 5, lane = tid & 31;
  const int wm = wave & 3, wn = wave >> 2;

  float bR[16];
  auto loadB = [&](int k0) {
    const float* s = ROUTED ? (Wd + (size_t)(k0 + bk) * HID + h0 + bn0)
                            : (Wd + (size_t)(h0 + bn) * FDIM + k0 + bkh);
#pragma unroll
    for (int i = 0; i < 16; ++i) bR[i] = s[i];
  };
  auto storeB = [&](int buf) {
    unsigned short* BL = Bs + buf * (128 * 32);
    if (ROUTED) {
#pragma unroll
      for (int i = 0; i < 16; ++i) BL[(bn0 + i) * 32 + bk] = f2bf_fast(bR[i]);
    } else {
      union { unsigned u[8]; uint4 q[2]; } t;
#pragma unroll
      for (int j = 0; j < 8; ++j) t.u[j] = pack2bf(bR[2 * j], bR[2 * j + 1]);
      uint4* dst = (uint4*)(BL + bn * 32 + bkh);
      dst[0] = t.q[0];
      dst[1] = t.q[1];
    }
  };

#if HAVE_ASYNC_LDS
  if (!avalid) {  // zero invalid rows once; async never touches them
    uint4 z = {0u, 0u, 0u, 0u};
    uint4* d0 = (uint4*)(As + arow * 32 + akh);
    uint4* d1 = (uint4*)(As + 128 * 32 + arow * 32 + akh);
    d0[0] = z; d0[1] = z;
    d1[0] = z; d1[1] = z;
  }
  auto issueA = [&](int k0, int buf) {
    if (avalid) {
      const unsigned short* g = aSrc + k0;
      const unsigned short* l = As + buf * (128 * 32) + arow * 32 + akh;
      async_b128(g, l);
      async_b128(g + 8, l + 8);
    }
  };
#else
  uint4 aR[2];
  auto loadA = [&](int k0) {
    if (avalid) {
      const uint4* s = (const uint4*)(aSrc + k0);
      aR[0] = s[0];
      aR[1] = s[1];
    } else {
      uint4 z = {0u, 0u, 0u, 0u};
      aR[0] = z;
      aR[1] = z;
    }
  };
  auto storeA = [&](int buf) {
    uint4* d = (uint4*)(As + buf * (128 * 32) + arow * 32 + akh);
    d[0] = aR[0];
    d[1] = aR[1];
  };
#endif

  const int NIT = FDIM / 32;
#if HAVE_ASYNC_LDS
  issueA(0, 0);
#else
  loadA(0);
#endif
  loadB(0);

  for (int i = 0; i < NIT; ++i) {
    const int buf = i & 1;
    storeB(buf);
#if HAVE_ASYNC_LDS
    wait_async0();   // A tile i resident in LDS before publishing barrier
#else
    storeA(buf);
#endif
    __syncthreads();
    if (i + 1 < NIT) {
#if HAVE_ASYNC_LDS
      issueA((i + 1) * 32, buf ^ 1);   // DMA overlaps the WMMAs below
#else
      loadA((i + 1) * 32);
#endif
      loadB((i + 1) * 32);
    }
    const unsigned short* Ab = As + buf * (128 * 32);
    const unsigned short* Bb = Bs + buf * (128 * 32);
    v16bf a0 = frag_a(Ab, wm * 32, lane);
    v16bf a1 = frag_a(Ab, wm * 32 + 16, lane);
#pragma unroll
    for (int ni = 0; ni < 4; ++ni) {
      v16bf b = frag_b(Bb, wn * 64 + ni * 16, lane);
      acc[ni] = WMMA_BF16(a0, b, acc[ni]);
      acc[4 + ni] = WMMA_BF16(a1, b, acc[4 + ni]);
    }
  }

  const int nloc = lane & 15;
  const int mb = (lane < 16) ? 0 : 8;
#pragma unroll
  for (int mi = 0; mi < 2; ++mi) {
#pragma unroll
    for (int ni = 0; ni < 4; ++ni) {
      v8f a = acc[mi * 4 + ni];
#pragma unroll
      for (int r = 0; r < 8; ++r) {
        const int m = wm * 32 + mi * 16 + mb + r;
        const int slot = m0 + m;
        if (slot < cnt) {
          const int tok = ROUTED ? tlist[base + slot] : slot;
          const int h = h0 + wn * 64 + ni * 16 + nloc;
          float* o = out + (size_t)tok * HID + h;
          if (ROUTED)
            *o += a[r];   // top-1: token rows are expert-disjoint
          else
            *o = a[r];    // shared expert writes first
        }
      }
    }
  }
}

// ---------------------------------------------------------------------------
extern "C" void kernel_launch(void* const* d_in, const int* in_sizes, int n_in,
                              void* d_out, int out_size, void* d_ws, size_t ws_size,
                              hipStream_t stream) {
  const float* X   = (const float*)d_in[0];  // (T,H)
  const float* Wr  = (const float*)d_in[1];  // (E,H)
  const float* Wgu = (const float*)d_in[2];  // (E,H,2F)
  const float* Wdn = (const float*)d_in[3];  // (E,F,H)
  const float* Wsg = (const float*)d_in[4];  // (F,H)
  const float* Wsu = (const float*)d_in[5];  // (F,H)
  const float* Wsd = (const float*)d_in[6];  // (H,F)

  float* out = (float*)d_out;                // (T,H)
  float* logits = out + (size_t)TOK * HID;   // (T,E)

  char* ws = (char*)d_ws;
  int* expert_id = (int*)ws;  ws += (size_t)TOK * 4;
  float* score = (float*)ws;  ws += (size_t)TOK * 4;
  int* counts = (int*)ws;     ws += NE * 4;
  int* offsets = (int*)ws;    ws += NE * 4;
  int* tlist = (int*)ws;      ws += (size_t)TOK * 4;
  size_t off = (size_t)(ws - (char*)d_ws);
  off = (off + 255) & ~(size_t)255;
  unsigned short* inter = (unsigned short*)((char*)d_ws + off);  // (T,F) bf16, reused

  router_kernel<<<TOK, 32, 0, stream>>>(X, Wr, logits, expert_id, score);
  count_kernel<<<1, 256, 0, stream>>>(expert_id, counts, offsets);
  rank_kernel<<<TOK / 256, 256, 0, stream>>>(expert_id, offsets, tlist);

  // Shared expert: SwiGLU -> out =
  moe_gateup<false><<<dim3(TOK / 128, FDIM / 64, 1), 256, 0, stream>>>(
      X, Wsg, Wsu, nullptr, nullptr, nullptr, nullptr, inter);
  moe_down<false><<<dim3(TOK / 128, HID / 128, 1), 256, 0, stream>>>(
      inter, Wsd, nullptr, nullptr, nullptr, out);

  // Routed experts (grouped GEMM, reuse inter): out +=
  moe_gateup<true><<<dim3(TOK / 128, FDIM / 64, NE), 256, 0, stream>>>(
      X, Wgu, nullptr, counts, offsets, tlist, score, inter);
  moe_down<true><<<dim3(TOK / 128, HID / 128, NE), 256, 0, stream>>>(
      inter, Wdn, counts, offsets, tlist, out);
}
